// Model_16844861734986
// MI455X (gfx1250) — matrix operand order repaired and checked by reading, not compiled, not run on hardware
//
#include <hip/hip_runtime.h>
#include <math.h>

#define BDIM 64
#define TDIM 512
#define CDIM 862
#define PRED 96
#define NEXP 4
#define DMODEL 256
#define NSER (BDIM*CDIM)          // 55168 (divisible by 16)
#define NFREQ 257
#define FCOLS 528                 // 2*257=514 padded to 16
#define CPAD 864                  // 862 padded to 16
#define CTILES 54                 // ceil(862/16)

typedef __bf16 bf16;
typedef __attribute__((ext_vector_type(16))) __bf16 v16bf;
typedef __attribute__((ext_vector_type(8)))  __bf16 v8bf;
typedef __attribute__((ext_vector_type(8)))  float  v8f;

// ---------------- WMMA helpers (CDNA5 16x16x32 bf16, f32 accum) ----------------

__device__ __forceinline__ v8f wmma_bf16(v16bf a, v16bf b, v8f c) {
  return __builtin_amdgcn_wmma_f32_16x16x32_bf16(false, a, false, b, (short)0, c, false, false);
}

// A tile 16x32, row-major source. Lane layout per ISA:
// lanes 0-15: row=lane, elems 0-7 -> K k0..k0+7, elems 8-15 -> K k0+16..k0+23
// lanes16-31: row=lane-16, elems 0-7 -> K k0+8..k0+15, elems 8-15 -> K k0+24..k0+31
__device__ __forceinline__ v16bf load_A_rm(const bf16* __restrict__ A, long lda,
                                           long row0, long k0, int rmax) {
  const int lane = threadIdx.x & 31, half = lane >> 4, l = lane & 15;
  v16bf a;
  if (l < rmax) {
    const bf16* p = A + (row0 + l) * lda + k0 + half * 8;
    v8bf lo = *(const v8bf*)p;
    v8bf hi = *(const v8bf*)(p + 16);
    __builtin_prefetch((const void*)(p + 32), 0, 1);   // global_prefetch next K-chunk
#pragma unroll
    for (int j = 0; j < 8; ++j) { a[j] = lo[j]; a[j + 8] = hi[j]; }
  } else {
#pragma unroll
    for (int j = 0; j < 16; ++j) a[j] = (bf16)0.0f;
  }
  return a;
}

// B tile 32x16, source stored COLUMN-major (BT[col][k], leading dim ldb over k):
// lanes 0-15: col=lane, elems 0-7 -> K k0..k0+7, elems 8-15 -> K k0+16..k0+23 ; lanes 16-31: col=lane-16, elems 0-7 -> K k0+8..k0+15, elems 8-15 -> K k0+24..k0+31
__device__ __forceinline__ v16bf load_B_cm(const bf16* __restrict__ BT, long ldb,
                                           long col0, long k0, int cmax) {
  const int lane = threadIdx.x & 31, half = lane >> 4, l = lane & 15;
  v16bf b;
  if (l < cmax) {
    const bf16* p = BT + (col0 + l) * ldb + k0 + half * 8;
    v8bf lo = *(const v8bf*)p;
    v8bf hi = *(const v8bf*)(p + 16);
#pragma unroll
    for (int j = 0; j < 8; ++j) { b[j] = lo[j]; b[j + 8] = hi[j]; }
  } else {
#pragma unroll
    for (int j = 0; j < 16; ++j) b[j] = (bf16)0.0f;
  }
  return b;
}

// ---------------- elementwise / reduction kernels ----------------

// RevIN stats + normalize + transpose [B,T,C] -> xs[N=T-major series]
__global__ void k_revin(const float* __restrict__ x, float* __restrict__ meanv,
                        float* __restrict__ stdv, bf16* __restrict__ xs) {
  __shared__ float s1[256], s2[256], mst[2];
  const long n = blockIdx.x;
  const long b = n / CDIM, c = n % CDIM;
  const int tid = threadIdx.x;
  const float* base = x + (long)b * TDIM * CDIM + c;
  float a0 = base[(long)tid * CDIM];
  float a1 = base[(long)(tid + 256) * CDIM];
  s1[tid] = a0 + a1; s2[tid] = a0 * a0 + a1 * a1;
  __syncthreads();
  for (int s = 128; s > 0; s >>= 1) {
    if (tid < s) { s1[tid] += s1[tid + s]; s2[tid] += s2[tid + s]; }
    __syncthreads();
  }
  if (tid == 0) {
    float m = s1[0] / (float)TDIM;
    float var = s2[0] / (float)TDIM - m * m;
    float sd = sqrtf(var + 1e-5f);
    meanv[n] = m; stdv[n] = sd;
    mst[0] = m; mst[1] = 1.0f / sd;
  }
  __syncthreads();
  const float m = mst[0], r = mst[1];
  xs[n * TDIM + tid]       = (bf16)((a0 - m) * r);
  xs[n * TDIM + tid + 256] = (bf16)((a1 - m) * r);
}

// series decomposition: season = x - moving_avg5(replicate pad)
__global__ void k_decomp(const bf16* __restrict__ xs, bf16* __restrict__ season) {
  const long idx = (long)blockIdx.x * 256 + threadIdx.x;
  const long n = idx >> 9; const int t = idx & 511;
  const bf16* row = xs + n * TDIM;
  float s = 0.f;
#pragma unroll
  for (int d = -2; d <= 2; ++d) {
    int tt = t + d; tt = tt < 0 ? 0 : (tt > 511 ? 511 : tt);
    s += (float)row[tt];
  }
  season[idx] = (bf16)((float)row[t] - s * 0.2f);
}

// rDFT basis, stored column-major: basisT[col][t]; col=2f -> cos, col=2f+1 -> -sin
__global__ void k_basis(bf16* __restrict__ basisT) {
  const long idx = (long)blockIdx.x * 256 + threadIdx.x;  // < FCOLS*TDIM
  const int col = (int)(idx >> 9), t = (int)(idx & 511);
  float v = 0.f;
  if (col < 2 * NFREQ) {
    const int f = col >> 1;
    const float ang = 6.283185307179586f * (float)f * (float)t * (1.0f / 512.0f);
    v = (col & 1) ? -__sinf(ang) : __cosf(ang);
  }
  basisT[idx] = (bf16)v;
}

// dft[N,528] = season[N,512] @ basisT'  (WMMA)
__global__ void k_gemm_dft(const bf16* __restrict__ season, const bf16* __restrict__ basisT,
                           float* __restrict__ dft) {
  const long CT = FCOLS / 16;
  const long tile = (long)blockIdx.x * 4 + (threadIdx.x >> 5);
  if (tile >= (long)(NSER / 16) * CT) return;
  const long rt = tile / CT, ct = tile % CT;
  const long row0 = rt * 16, col0 = ct * 16;
  v8f acc = {};
  for (int k0 = 0; k0 < TDIM; k0 += 32)
    acc = wmma_bf16(load_A_rm(season, TDIM, row0, k0, 16),
                    load_B_cm(basisT, TDIM, col0, k0, 16), acc);
  const int lane = threadIdx.x & 31, half = lane >> 4, l = lane & 15;
#pragma unroll
  for (int r = 0; r < 8; ++r)
    dft[(row0 + r + 8 * half) * FCOLS + col0 + l] = acc[r];
}

// ampC[b,f] = mean_c |dft(b,c,f)|
__global__ void k_amp(const float* __restrict__ dft, float* __restrict__ ampC) {
  __shared__ float sm[256];
  const int b = blockIdx.x / NFREQ, f = blockIdx.x % NFREQ;
  const int tid = threadIdx.x;
  float s = 0.f;
  for (int c = tid; c < CDIM; c += 256) {
    const long n = (long)b * CDIM + c;
    const float re = dft[n * FCOLS + 2 * f];
    const float im = dft[n * FCOLS + 2 * f + 1];
    s += sqrtf(re * re + im * im);
  }
  sm[tid] = s; __syncthreads();
  for (int st = 128; st > 0; st >>= 1) { if (tid < st) sm[tid] += sm[tid + st]; __syncthreads(); }
  if (tid == 0) ampC[b * NFREQ + f] = sm[0] / (float)CDIM;
}

// freq mean over B, top-2 freqs, fft_bias[b,e] = period_weight[b,:] @ w_fft
__global__ void k_topk(const float* __restrict__ ampC, const float* __restrict__ w_fft,
                       float* __restrict__ fft_bias) {
  __shared__ float fr[NFREQ];
  __shared__ int topi[2];
  const int tid = threadIdx.x;
  for (int f = tid; f < NFREQ; f += 256) {
    float s = 0.f;
    for (int b = 0; b < BDIM; ++b) s += ampC[b * NFREQ + f];
    fr[f] = (f == 0) ? 0.f : s / (float)BDIM;
  }
  __syncthreads();
  if (tid == 0) {
    int i1 = 0; float v1 = -1e30f;
    for (int f = 0; f < NFREQ; ++f) if (fr[f] > v1) { v1 = fr[f]; i1 = f; }
    int i2 = 0; float v2 = -1e30f;
    for (int f = 0; f < NFREQ; ++f) if (f != i1 && fr[f] > v2) { v2 = fr[f]; i2 = f; }
    topi[0] = i1; topi[1] = i2;
  }
  __syncthreads();
  const int b = tid >> 2, e = tid & 3;   // 64*4 == 256
  const float pw0 = ampC[b * NFREQ + topi[0]];
  const float pw1 = ampC[b * NFREQ + topi[1]];
  fft_bias[b * NEXP + e] = pw0 * w_fft[0 * NEXP + e] + pw1 * w_fft[1 * NEXP + e];
}

__global__ void k_zero8(float* __restrict__ p) { if (threadIdx.x < 8) p[threadIdx.x] = 0.f; }

// noisy-top-k gating (eval): logits = xs@w_gate + fft_bias; top-2 softmax; importance/load atomics
__global__ void k_gate(const bf16* __restrict__ xs, const float* __restrict__ w_gate,
                       const float* __restrict__ fft_bias, float* __restrict__ gates,
                       float* __restrict__ imp) {
  __shared__ float lg[64][4];
  const int tid = threadIdx.x;
  const int local = tid >> 2, e = tid & 3;
  const long n = (long)blockIdx.x * 64 + local;
  const long b = n / CDIM;
  const bf16* row = xs + n * TDIM;
  float acc = 0.f;
  for (int t = 0; t < TDIM; ++t) acc += (float)row[t] * w_gate[t * NEXP + e];
  lg[local][e] = acc + fft_bias[b * NEXP + e];
  __syncthreads();
  if (e == 0) {
    float v[4];
#pragma unroll
    for (int i = 0; i < 4; ++i) v[i] = lg[local][i];
    int i1 = 0;
    for (int i = 1; i < 4; ++i) if (v[i] > v[i1]) i1 = i;
    int i2 = -1;
    for (int i = 0; i < 4; ++i) if (i != i1 && (i2 < 0 || v[i] > v[i2])) i2 = i;
    const float e2 = __expf(v[i2] - v[i1]);
    const float inv = 1.0f / (1.0f + e2);
    const float g1 = inv, g2 = e2 * inv;
    for (int i = 0; i < 4; ++i)
      gates[n * 4 + i] = (i == i1) ? g1 : ((i == i2) ? g2 : 0.f);
    atomicAdd(&imp[i1], g1);       atomicAdd(&imp[i2], g2);
    atomicAdd(&imp[4 + i1], 1.0f); atomicAdd(&imp[4 + i2], (g2 > 0.f) ? 1.0f : 0.f);
  }
}

// moe[n,p] = sum_e gates[n,e]*(xs[n,:]@W_e[:,p] + b_e[p])   (E folded into K loop)
__global__ void k_gemm_moe(const bf16* __restrict__ xs, const bf16* __restrict__ eWT,
                           const float* __restrict__ gates, const float* __restrict__ expert_b,
                           float* __restrict__ moe) {
  const long CT = PRED / 16;
  const long tile = (long)blockIdx.x * 4 + (threadIdx.x >> 5);
  if (tile >= (long)(NSER / 16) * CT) return;
  const long rt = tile / CT, ct = tile % CT;
  const long row0 = rt * 16, col0 = ct * 16;
  const int lane = threadIdx.x & 31, half = lane >> 4, l = lane & 15;
  const int pcol = (int)col0 + l;
  v8f fin = {};
  for (int e = 0; e < NEXP; ++e) {
    v8f acc = {};
    const bf16* BT = eWT + (long)e * PRED * TDIM;
    for (int k0 = 0; k0 < TDIM; k0 += 32)
      acc = wmma_bf16(load_A_rm(xs, TDIM, row0, k0, 16),
                      load_B_cm(BT, TDIM, col0, k0, 16), acc);
    const float be = expert_b[e * PRED + pcol];
#pragma unroll
    for (int r = 0; r < 8; ++r) {
      const long n = row0 + r + 8 * half;
      fin[r] += gates[n * 4 + e] * (acc[r] + be);
    }
  }
#pragma unroll
  for (int r = 0; r < 8; ++r)
    moe[(row0 + r + 8 * half) * PRED + pcol] = fin[r];
}

// generic f32 [K,Ncols] -> bf16 transposed [Ncols,K]
__global__ void k_transpose_bf16(const float* __restrict__ src, bf16* __restrict__ dst,
                                 int K, int Ncols) {
  const long idx = (long)blockIdx.x * 256 + threadIdx.x;
  if (idx >= (long)K * Ncols) return;
  const long j = idx / K, k = idx % K;
  dst[j * K + k] = (bf16)src[k * Ncols + j];
}

// q/k row-major [N,256]; v stored transposed per batch vT[b][dm][CPAD]
__global__ void k_gemm_qkv(const bf16* __restrict__ xs, const bf16* __restrict__ wT,
                           bf16* __restrict__ qb, bf16* __restrict__ kb, bf16* __restrict__ vT) {
  const int z = blockIdx.z;
  const long CT = DMODEL / 16;
  const long tile = (long)blockIdx.x * 4 + (threadIdx.x >> 5);
  if (tile >= (long)(NSER / 16) * CT) return;
  const long rt = tile / CT, ct = tile % CT;
  const long row0 = rt * 16, col0 = ct * 16;
  const bf16* BT = wT + (long)z * DMODEL * TDIM;
  v8f acc = {};
  for (int k0 = 0; k0 < TDIM; k0 += 32)
    acc = wmma_bf16(load_A_rm(xs, TDIM, row0, k0, 16),
                    load_B_cm(BT, TDIM, col0, k0, 16), acc);
  const int lane = threadIdx.x & 31, half = lane >> 4, l = lane & 15;
  if (z < 2) {
    bf16* dst = (z == 0) ? qb : kb;
#pragma unroll
    for (int r = 0; r < 8; ++r)
      dst[(row0 + r + 8 * half) * DMODEL + col0 + l] = (bf16)acc[r];
  } else {
#pragma unroll
    for (int r = 0; r < 8; ++r) {
      const long n = row0 + r + 8 * half;
      const long b = n / CDIM, c = n % CDIM;
      vT[((long)b * DMODEL + col0 + l) * CPAD + c] = (bf16)acc[r];
    }
  }
}

// S[b,c,e] = (Q Kt)*scale ; K row-major == column-major B for QK^T
__global__ void k_attn_score(const bf16* __restrict__ qb, const bf16* __restrict__ kb,
                             float* __restrict__ S) {
  const long tile = (long)blockIdx.x * 4 + (threadIdx.x >> 5);
  if (tile >= (long)CTILES * CTILES) return;
  const int b = blockIdx.y;
  const long rt = tile / CTILES, ct = tile % CTILES;
  int rmax = CDIM - (int)rt * 16; rmax = rmax > 16 ? 16 : rmax;
  int cmax = CDIM - (int)ct * 16; cmax = cmax > 16 ? 16 : cmax;
  const bf16* A  = qb + (long)b * CDIM * DMODEL;
  const bf16* BT = kb + (long)b * CDIM * DMODEL;
  v8f acc = {};
  for (int k0 = 0; k0 < DMODEL; k0 += 32)
    acc = wmma_bf16(load_A_rm(A, DMODEL, rt * 16, k0, rmax),
                    load_B_cm(BT, DMODEL, ct * 16, k0, cmax), acc);
  const int lane = threadIdx.x & 31, half = lane >> 4, l = lane & 15;
  const float scale = 0.0625f;   // 1/sqrt(DM=256)
#pragma unroll
  for (int r = 0; r < 8; ++r) {
    const int row = (int)rt * 16 + r + 8 * half;
    if (row < CDIM)
      S[((long)b * CDIM + row) * CPAD + ct * 16 + l] = acc[r] * scale;
  }
}

// row softmax over 862 valid cols; write bf16 probs with zero pad cols
__global__ void k_softmax(const float* __restrict__ S, bf16* __restrict__ P) {
  __shared__ float sm[256];
  const long row = blockIdx.x;
  const float* src = S + row * CPAD;
  bf16* dst = P + row * CPAD;
  const int tid = threadIdx.x;
  float m = -1e30f;
  for (int i = tid; i < CDIM; i += 256) m = fmaxf(m, src[i]);
  sm[tid] = m; __syncthreads();
  for (int s = 128; s > 0; s >>= 1) { if (tid < s) sm[tid] = fmaxf(sm[tid], sm[tid + s]); __syncthreads(); }
  m = sm[0]; __syncthreads();
  float sum = 0.f;
  for (int i = tid; i < CDIM; i += 256) sum += __expf(src[i] - m);
  sm[tid] = sum; __syncthreads();
  for (int s = 128; s > 0; s >>= 1) { if (tid < s) sm[tid] += sm[tid + s]; __syncthreads(); }
  const float inv = 1.0f / sm[0];
  for (int i = tid; i < CDIM; i += 256) dst[i] = (bf16)(__expf(src[i] - m) * inv);
  if (tid < CPAD - CDIM) dst[CDIM + tid] = (bf16)0.f;
}

// O[b,c,dm] = P[b] @ V[b]  (K = CPAD, pad cols have P==0)
__global__ void k_gemm_attnO(const bf16* __restrict__ P, const bf16* __restrict__ vT,
                             bf16* __restrict__ O) {
  const long CT = DMODEL / 16;
  const long tile = (long)blockIdx.x * 4 + (threadIdx.x >> 5);
  if (tile >= (long)CTILES * CT) return;
  const int b = blockIdx.y;
  const long rt = tile / CT, ct = tile % CT;
  int rmax = CDIM - (int)rt * 16; rmax = rmax > 16 ? 16 : rmax;
  const bf16* A  = P  + (long)b * CDIM * CPAD;
  const bf16* BT = vT + (long)b * DMODEL * CPAD;
  v8f acc = {};
  for (int k0 = 0; k0 < CPAD; k0 += 32)
    acc = wmma_bf16(load_A_rm(A, CPAD, rt * 16, k0, rmax),
                    load_B_cm(BT, CPAD, ct * 16, k0, 16), acc);
  const int lane = threadIdx.x & 31, half = lane >> 4, l = lane & 15;
#pragma unroll
  for (int r = 0; r < 8; ++r) {
    const int c = (int)rt * 16 + r + 8 * half;
    if (c < CDIM)
      O[((long)b * CDIM + c) * DMODEL + ct * 16 + l] = (bf16)acc[r];
  }
}

// var_out[n,p] = attn_o @ out_W
__global__ void k_gemm_varout(const bf16* __restrict__ O, const bf16* __restrict__ outWT,
                              float* __restrict__ varout) {
  const long CT = PRED / 16;
  const long tile = (long)blockIdx.x * 4 + (threadIdx.x >> 5);
  if (tile >= (long)(NSER / 16) * CT) return;
  const long rt = tile / CT, ct = tile % CT;
  const long row0 = rt * 16, col0 = ct * 16;
  v8f acc = {};
  for (int k0 = 0; k0 < DMODEL; k0 += 32)
    acc = wmma_bf16(load_A_rm(O, DMODEL, row0, k0, 16),
                    load_B_cm(outWT, DMODEL, col0, k0, 16), acc);
  const int lane = threadIdx.x & 31, half = lane >> 4, l = lane & 15;
#pragma unroll
  for (int r = 0; r < 8; ++r)
    varout[(row0 + r + 8 * half) * PRED + col0 + l] = acc[r];
}

// A loader for the head: concat [moe | var] along K (f32 -> bf16 on the fly)
__device__ __forceinline__ v16bf load_A_head(const float* __restrict__ moe,
                                             const float* __restrict__ var,
                                             long row0, int k0) {
  const int lane = threadIdx.x & 31, half = lane >> 4, l = lane & 15;
  const long r = row0 + l;
  v16bf a;
#pragma unroll
  for (int j = 0; j < 8; ++j) {
    const int k1 = k0 + half * 8 + j;
    const int k2 = k0 + 16 + half * 8 + j;
    const float f1 = (k1 < PRED) ? moe[r * PRED + k1] : var[r * PRED + k1 - PRED];
    const float f2 = (k2 < PRED) ? moe[r * PRED + k2] : var[r * PRED + k2 - PRED];
    a[j] = (bf16)f1; a[j + 8] = (bf16)f2;
  }
  return a;
}

// head GEMM + bias + RevIN denorm + transpose to [B,P,C]
__global__ void k_gemm_head(const float* __restrict__ moe, const float* __restrict__ var,
                            const bf16* __restrict__ headWT, const float* __restrict__ head_b,
                            const float* __restrict__ meanv, const float* __restrict__ stdv,
                            float* __restrict__ out) {
  const long CT = PRED / 16;
  const long tile = (long)blockIdx.x * 4 + (threadIdx.x >> 5);
  if (tile >= (long)(NSER / 16) * CT) return;
  const long rt = tile / CT, ct = tile % CT;
  const long row0 = rt * 16, col0 = ct * 16;
  v8f acc = {};
  for (int k0 = 0; k0 < 2 * PRED; k0 += 32)
    acc = wmma_bf16(load_A_head(moe, var, row0, k0),
                    load_B_cm(headWT, 2 * PRED, col0, k0, 16), acc);
  const int lane = threadIdx.x & 31, half = lane >> 4, l = lane & 15;
  const int p = (int)col0 + l;
  const float hb = head_b[p];
#pragma unroll
  for (int r = 0; r < 8; ++r) {
    const long n = row0 + r + 8 * half;
    const long b = n / CDIM, c = n % CDIM;
    out[(b * PRED + p) * CDIM + c] = (acc[r] + hb) * stdv[n] + meanv[n];
  }
}

// balance loss = cv2(importance) + cv2(load)
__global__ void k_balance(const float* __restrict__ imp, float* __restrict__ out_loss) {
  if (threadIdx.x == 0) {
    float mi = 0.f, ml = 0.f;
    for (int i = 0; i < 4; ++i) { mi += imp[i]; ml += imp[4 + i]; }
    mi *= 0.25f; ml *= 0.25f;
    float vi = 0.f, vl = 0.f;
    for (int i = 0; i < 4; ++i) {
      const float d = imp[i] - mi;     vi += d * d;
      const float e = imp[4 + i] - ml; vl += e * e;
    }
    vi *= 0.25f; vl *= 0.25f;
    out_loss[0] = vi / (mi * mi + 1e-10f) + vl / (ml * ml + 1e-10f);
  }
}

// ---------------- host ----------------

extern "C" void kernel_launch(void* const* d_in, const int* in_sizes, int n_in,
                              void* d_out, int out_size, void* d_ws, size_t ws_size,
                              hipStream_t stream) {
  (void)in_sizes; (void)n_in; (void)out_size; (void)ws_size;
  const float* x_enc    = (const float*)d_in[0];
  const float* w_gate   = (const float*)d_in[4];
  const float* w_fft    = (const float*)d_in[5];
  const float* expert_W = (const float*)d_in[6];
  const float* expert_b = (const float*)d_in[7];
  const float* q_W      = (const float*)d_in[8];
  const float* k_W      = (const float*)d_in[9];
  const float* v_W      = (const float*)d_in[10];
  const float* out_W    = (const float*)d_in[11];
  const float* head_W   = (const float*)d_in[12];
  const float* head_b   = (const float*)d_in[13];
  float* out = (float*)d_out;

  char* w = (char*)d_ws; size_t off = 0;
  auto alloc = [&](size_t bytes) -> void* {
    void* p = w + off; off = (off + bytes + 255) & ~(size_t)255; return p;
  };
  float* meanv  = (float*)alloc((size_t)NSER * 4);
  float* stdv   = (float*)alloc((size_t)NSER * 4);
  bf16*  xs     = (bf16*) alloc((size_t)NSER * TDIM * 2);
  bf16*  season = (bf16*) alloc((size_t)NSER * TDIM * 2);   // later aliased as attn_o
  bf16*  attn_o = season;                                    // dead after DFT GEMM
  void*  big0   = alloc((size_t)BDIM * CDIM * CPAD * 4);     // dft (116MB) then S (190MB)
  float* dft    = (float*)big0;
  float* S      = (float*)big0;
  float* ampC   = (float*)alloc((size_t)BDIM * NFREQ * 4);
  float* fftb   = (float*)alloc((size_t)BDIM * NEXP * 4);
  float* gates  = (float*)alloc((size_t)NSER * NEXP * 4);
  float* imp    = (float*)alloc(64);
  bf16*  basisT = (bf16*) alloc((size_t)FCOLS * TDIM * 2);
  bf16*  eWT    = (bf16*) alloc((size_t)NEXP * PRED * TDIM * 2);
  bf16*  qkvWT  = (bf16*) alloc((size_t)3 * DMODEL * TDIM * 2);
  bf16*  outWT  = (bf16*) alloc((size_t)PRED * DMODEL * 2);
  bf16*  headWT = (bf16*) alloc((size_t)PRED * 2 * PRED * 2);
  bf16*  qb     = (bf16*) alloc((size_t)NSER * DMODEL * 2);
  bf16*  kb     = (bf16*) alloc((size_t)NSER * DMODEL * 2);
  bf16*  vT     = (bf16*) alloc((size_t)BDIM * DMODEL * CPAD * 2);
  bf16*  Pm     = (bf16*) alloc((size_t)BDIM * CDIM * CPAD * 2);
  float* moe    = (float*)alloc((size_t)NSER * PRED * 4);
  float* varo   = (float*)alloc((size_t)NSER * PRED * 4);

  auto tgrid = [](long total) { return dim3((unsigned)((total + 255) / 256)); };

  // weight transposition/conversion to bf16 column-major
  for (int e = 0; e < NEXP; ++e)
    k_transpose_bf16<<<tgrid((long)TDIM * PRED), 256, 0, stream>>>(
        expert_W + (long)e * TDIM * PRED, eWT + (long)e * PRED * TDIM, TDIM, PRED);
  k_transpose_bf16<<<tgrid((long)TDIM * DMODEL), 256, 0, stream>>>(q_W, qkvWT + 0L * DMODEL * TDIM, TDIM, DMODEL);
  k_transpose_bf16<<<tgrid((long)TDIM * DMODEL), 256, 0, stream>>>(k_W, qkvWT + 1L * DMODEL * TDIM, TDIM, DMODEL);
  k_transpose_bf16<<<tgrid((long)TDIM * DMODEL), 256, 0, stream>>>(v_W, qkvWT + 2L * DMODEL * TDIM, TDIM, DMODEL);
  k_transpose_bf16<<<tgrid((long)DMODEL * PRED), 256, 0, stream>>>(out_W, outWT, DMODEL, PRED);
  k_transpose_bf16<<<tgrid((long)2 * PRED * PRED), 256, 0, stream>>>(head_W, headWT, 2 * PRED, PRED);

  // pipeline
  k_revin<<<NSER, 256, 0, stream>>>(x_enc, meanv, stdv, xs);
  k_decomp<<<(NSER * TDIM) / 256, 256, 0, stream>>>(xs, season);
  k_basis<<<(FCOLS * TDIM) / 256, 256, 0, stream>>>(basisT);
  { long tiles = (long)(NSER / 16) * (FCOLS / 16);
    k_gemm_dft<<<(unsigned)((tiles + 3) / 4), 128, 0, stream>>>(season, basisT, dft); }
  k_amp<<<BDIM * NFREQ, 256, 0, stream>>>(dft, ampC);
  k_topk<<<1, 256, 0, stream>>>(ampC, w_fft, fftb);
  k_zero8<<<1, 32, 0, stream>>>(imp);
  k_gate<<<NSER / 64, 256, 0, stream>>>(xs, w_gate, fftb, gates, imp);
  { long tiles = (long)(NSER / 16) * (PRED / 16);
    k_gemm_moe<<<(unsigned)((tiles + 3) / 4), 128, 0, stream>>>(xs, eWT, gates, expert_b, moe); }
  { long tiles = (long)(NSER / 16) * (DMODEL / 16);
    dim3 g((unsigned)((tiles + 3) / 4), 1, 3);
    k_gemm_qkv<<<g, 128, 0, stream>>>(xs, qkvWT, qb, kb, vT); }
  { long tiles = (long)CTILES * CTILES;
    dim3 g((unsigned)((tiles + 3) / 4), BDIM);
    k_attn_score<<<g, 128, 0, stream>>>(qb, kb, S); }
  k_softmax<<<BDIM * CDIM, 256, 0, stream>>>(S, Pm);
  { long tiles = (long)CTILES * (DMODEL / 16);
    dim3 g((unsigned)((tiles + 3) / 4), BDIM);
    k_gemm_attnO<<<g, 128, 0, stream>>>(Pm, vT, attn_o); }
  { long tiles = (long)(NSER / 16) * (PRED / 16);
    k_gemm_varout<<<(unsigned)((tiles + 3) / 4), 128, 0, stream>>>(attn_o, outWT, varo); }
  { long tiles = (long)(NSER / 16) * (PRED / 16);
    k_gemm_head<<<(unsigned)((tiles + 3) / 4), 128, 0, stream>>>(moe, varo, headWT, head_b,
                                                                meanv, stdv, out); }
  k_balance<<<1, 64, 0, stream>>>(imp, out + (long)BDIM * PRED * CDIM);
}
